// SBCoreMiniLM_85383949845398
// MI455X (gfx1250) — compile-verified, matrix operand order checked by reading, not run on hardware
//
#include <hip/hip_runtime.h>

// ---------------------------------------------------------------------------
// SBCoreMiniLM on MI455X (gfx1250, wave32, WMMA 16x16x32 f16)
//   K1: pack fp32 weights -> f16 WMMA-B operand layout (in d_ws, NT stores)
//   K2: embedding + pos + input projection (WMMA)  -> xh f16 [2048,128]
//   K3: persistent single-workgroup recurrent scan (WMMA + LDS state +
//       global_prefetch_b8 to hide L2 latency in the serial chain)
//   K4: head GEMM [2048,128]@[128,32000] (WMMA, M=64/N=128, NT output stores)
// ---------------------------------------------------------------------------

typedef __attribute__((ext_vector_type(16))) _Float16 v16h;
typedef __attribute__((ext_vector_type(8)))  float    v8f;
typedef __attribute__((ext_vector_type(4)))  unsigned int v4u;

#define WMMA_F16(a, b, c) \
  __builtin_amdgcn_wmma_f32_16x16x32_f16(false, (a), false, (b), (short)0, (c), false, false)

// 16-bit operand lane layout (16x16x32): half j of a lane maps to k offset:
//   lanes 0-15 base k=0, lanes 16-31 base k=8; vgpr pairs 0-3 -> k 0..7, 4-7 -> +16
__device__ __forceinline__ int kmap(int lane, int j) {
  int vg = j >> 1;
  return ((lane & 16) >> 1) + ((vg & 3) << 1) + (j & 1) + ((vg >> 2) << 4);
}

__device__ __forceinline__ float wave_sum(float v) {
#pragma unroll
  for (int off = 16; off > 0; off >>= 1) v += __shfl_xor(v, off, 32);
  return v;
}

// ---------------------------------------------------------------------------
// K1: pack W[Kdim,N] fp32 row-major -> f16 blocks of 512 halves per (ntile,kc):
//     dst[((nt*KC+kc)*32 + lane)*16 + j] = W[(kc*32 + kmap(lane,j))*N + nt*16 + lane%16]
// ---------------------------------------------------------------------------
__global__ void pack_wmma_b_k(const float* __restrict__ W, _Float16* __restrict__ dst,
                              int Kdim, int N) {
  int KC = Kdim >> 5;
  long total = (long)Kdim * N;
  for (long idx = (long)blockIdx.x * blockDim.x + threadIdx.x; idx < total;
       idx += (long)gridDim.x * blockDim.x) {
    int  j    = (int)(idx & 15);
    int  lane = (int)((idx >> 4) & 31);
    long blk  = idx >> 9;
    int  kc   = (int)(blk % KC);
    int  nt   = (int)(blk / KC);
    int  k    = kc * 32 + kmap(lane, j);
    int  n    = nt * 16 + (lane & 15);
    __builtin_nontemporal_store((_Float16)W[(long)k * N + n], dst + idx);
  }
}

// ---------------------------------------------------------------------------
// Wave-cooperative GEMM: out[8,128] = A[16,Kdim] @ Wp + bias, A rows 8..15 zero.
// One wave per 16-col tile (8 waves cover N=128). A in LDS (stride 384), f16.
// ---------------------------------------------------------------------------
__device__ __forceinline__ void gemm8(const _Float16* __restrict__ Wp,
                                      const _Float16* __restrict__ Ab, int KC,
                                      float* __restrict__ outb, const float* __restrict__ bias,
                                      int lane, int wave) {
  v8f acc = {};
  const _Float16* arow = Ab + (lane & 15) * 384;
  const _Float16* wp   = Wp + (size_t)(wave * KC) * 512 + lane * 16;
  for (int kc = 0; kc < KC; ++kc) {
    v16h a;
    const _Float16* ap = arow + kc * 32;
#pragma unroll
    for (int j = 0; j < 16; ++j) a[j] = ap[kmap(lane, j)];
    v16h b = *(const v16h*)(wp + kc * 512);
    acc = WMMA_F16(a, b, acc);
  }
  if (lane < 16) {  // rows M=0..7 live in lanes 0..15 (rows 8..15 are padding)
    int col = wave * 16 + lane;
    float bvv = bias ? bias[col] : 0.0f;
#pragma unroll
    for (int r = 0; r < 8; ++r) outb[r * 128 + col] = acc[r] + bvv;
  }
}

// Prefetch one layer-slice of packed weights (nh halves) into cache hierarchy.
__device__ __forceinline__ void prefetch_w(const _Float16* __restrict__ p, int nh, int tid) {
  for (int c = tid; c < (nh >> 7); c += 256)        // 128 halves = 256B cacheline
    __builtin_prefetch(p + (size_t)c * 128, 0, 0);  // -> global_prefetch_b8
}

// ---------------------------------------------------------------------------
// K2: x = (emb[ids] + pos) @ Win + b_in  -> xh f16 [2048,128]
// ---------------------------------------------------------------------------
__global__ void embed_in_k(const int* __restrict__ ids, const float* __restrict__ emb,
                           const float* __restrict__ pos, const _Float16* __restrict__ WinP,
                           const float* __restrict__ b_in, _Float16* __restrict__ xh) {
  __shared__ _Float16 As[16 * 128];
  int tid = threadIdx.x, lane = tid & 31, wave = tid >> 5;
  int mt = blockIdx.x;  // 0..127, 16 rows each
  for (int i = tid; i < 16 * 128; i += 256) {
    int r = i >> 7, d = i & 127;
    int rr = mt * 16 + r;
    int b = rr >> 8, t = rr & 255;
    float v = 0.0f;
    if (d < 96) {
      int id = ids[b * 256 + t];
      v = emb[(long)id * 96 + d] + pos[t * 96 + d];
    }
    As[r * 128 + d] = (_Float16)v;
  }
  __syncthreads();
  v8f acc = {};
  const _Float16* arow = As + (lane & 15) * 128;
  for (int kc = 0; kc < 3; ++kc) {  // K = 96
    v16h a;
#pragma unroll
    for (int j = 0; j < 16; ++j) a[j] = arow[kc * 32 + kmap(lane, j)];
    v16h b = *(const v16h*)(WinP + (size_t)(wave * 3 + kc) * 512 + lane * 16);
    acc = WMMA_F16(a, b, acc);
  }
  int n = wave * 16 + (lane & 15);
  int mbase = (lane & 16) ? 8 : 0;
  float bv = b_in[n];
#pragma unroll
  for (int r = 0; r < 8; ++r) {
    int row = mt * 16 + mbase + r;
    xh[(long)row * 128 + n] = (_Float16)(acc[r] + bv);
  }
}

// ---------------------------------------------------------------------------
// K3: persistent recurrent scan. grid(1) x block(256) = 8 waves. ~299 KB LDS.
// ---------------------------------------------------------------------------
__global__ void __launch_bounds__(256, 1)
scan_k(const _Float16* __restrict__ xh,
       const _Float16* __restrict__ WqP, const _Float16* __restrict__ WgP,
       const _Float16* __restrict__ WcP, const _Float16* __restrict__ WoP,
       const _Float16* __restrict__ WkP, const _Float16* __restrict__ WvP,
       const float* __restrict__ bg, const float* __restrict__ bc,
       const float* __restrict__ bo, const float* __restrict__ lng,
       const float* __restrict__ lnb, const float* __restrict__ bk,
       const float* __restrict__ bv, const float* __restrict__ Wwg,
       const float* __restrict__ bwg, const float* __restrict__ sem_k,
       const float* __restrict__ sem_v, const float* __restrict__ fng,
       const float* __restrict__ fnb, _Float16* __restrict__ hseq) {
  extern __shared__ char smem[];
  _Float16* wk   = (_Float16*)smem;          // [3][8][16][128] f16 working keys
  _Float16* wv   = wk + 3 * 8 * 16 * 128;    // [3][8][16][128] f16 working values
  _Float16* semk = wv + 3 * 8 * 16 * 128;    // [3][64][128] f16 normalized sem keys
  float* hid  = (float*)(semk + 3 * 64 * 128);  // [3][8][128]
  float* cur  = hid + 3 * 1024;                 // [8][128]
  _Float16* Ab = (_Float16*)(cur + 1024);       // [16][384] A staging (rows 8..15 zero)
  float* qb   = (float*)(Ab + 16 * 384);        // [8][128]
  float* sc   = qb + 1024;                      // [8][80] scores
  int*   topi = (int*)(sc + 640);               // [8][4]
  float* topw = (float*)(topi + 32);            // [8][4]
  float* memb = topw + 32;                      // [8][128]
  float* gb   = memb + 1024;
  float* cb   = gb + 1024;
  float* pb   = cb + 1024;
  float* nkb  = pb + 1024;
  float* nvb  = nkb + 1024;
  float* gate = nvb + 1024;                     // [8]

  const int tid = threadIdx.x, lane = tid & 31, wave = tid >> 5;

  // ---- init state ----
  for (int i = tid; i < (3 * 8 * 16 * 128) / 2; i += 256) {
    ((unsigned*)wk)[i] = 0u; ((unsigned*)wv)[i] = 0u;
  }
  for (int i = tid; i < 3 * 1024; i += 256) hid[i] = 0.0f;
  for (int i = tid; i < (16 * 384) / 2; i += 256) ((unsigned*)Ab)[i] = 0u;
  for (int row = tid; row < 3 * 64; row += 256) {  // l2norm(sem_k) -> LDS f16
    float s2 = 0.0f;
    for (int d = 0; d < 128; ++d) { float v = sem_k[row * 128 + d]; s2 += v * v; }
    float s = 1.0f / fmaxf(sqrtf(s2), 1e-6f);
    for (int d = 0; d < 128; ++d) semk[row * 128 + d] = (_Float16)(sem_k[row * 128 + d] * s);
  }
  __syncthreads();

  for (int t = 0; t < 256; ++t) {
    for (int i = tid; i < 1024; i += 256)
      cur[i] = (float)xh[(((i >> 7) << 8) + t) * 128 + (i & 127)];
    // warm the Wq slice for layer 0 while the loads above are in flight
    prefetch_w(WqP, 32768, tid);
    __syncthreads();

    for (int l = 0; l < 3; ++l) {
      const float* prev = hid + l * 1024;
      // A = [cur | prev]
      for (int i = tid; i < 1024; i += 256) {
        int b = i >> 7, d = i & 127;
        Ab[b * 384 + d]       = (_Float16)cur[i];
        Ab[b * 384 + 128 + d] = (_Float16)prev[i];
      }
      __syncthreads();
      gemm8(WqP + l * 32768, Ab, 8, qb, nullptr, lane, wave);  // q_raw
      __syncthreads();
      {  // l2norm(q), wave per row
        float x0 = qb[wave * 128 + lane],      x1 = qb[wave * 128 + lane + 32],
              x2 = qb[wave * 128 + lane + 64], x3 = qb[wave * 128 + lane + 96];
        float s2 = wave_sum(x0 * x0 + x1 * x1 + x2 * x2 + x3 * x3);
        float s = 1.0f / fmaxf(sqrtf(s2), 1e-6f);
        qb[wave * 128 + lane] = x0 * s;      qb[wave * 128 + lane + 32] = x1 * s;
        qb[wave * 128 + lane + 64] = x2 * s; qb[wave * 128 + lane + 96] = x3 * s;
      }
      // overlap: pull this layer's gate/candidate weights toward the WGP while
      // the router (pure LDS/VALU work) runs
      prefetch_w(WgP + l * 49152, 49152, tid);
      prefetch_w(WcP + l * 49152, 49152, tid);
      __syncthreads();
      // scores: working (0..15) + semantic (16..79)
      if (tid < 128) {
        int b = tid >> 4, i = tid & 15;
        const _Float16* kp = wk + ((l * 8 + b) * 16 + i) * 128;
        float dot = 0.0f, s2 = 0.0f;
        for (int d = 0; d < 128; ++d) {
          float kv = (float)kp[d];
          s2 += kv * kv; dot += qb[b * 128 + d] * kv;
        }
        sc[b * 80 + i] = dot / fmaxf(sqrtf(s2), 1e-6f);
      }
      for (int idx = tid; idx < 512; idx += 256) {
        int b = idx >> 6, j = idx & 63;
        const _Float16* kp = semk + (l * 64 + j) * 128;
        float dot = 0.0f;
        for (int d = 0; d < 128; ++d) dot += qb[b * 128 + d] * (float)kp[d];
        sc[b * 80 + 16 + j] = dot;
      }
      __syncthreads();
      // top-4 + softmax (one thread per batch row)
      if (tid < 8) {
        int b = tid, ti[4]; float tv[4];
        for (int k = 0; k < 4; ++k) {
          float best = -3.0e38f; int bi = 0;
          for (int i = 0; i < 80; ++i) {
            bool skip = false;
            for (int p = 0; p < k; ++p) skip |= (ti[p] == i);
            float v = sc[b * 80 + i];
            if (!skip && v > best) { best = v; bi = i; }
          }
          ti[k] = bi; tv[k] = best;
        }
        float s = 0.0f, w[4];
        for (int k = 0; k < 4; ++k) { w[k] = __expf(tv[k] - tv[0]); s += w[k]; }
        float inv = 1.0f / s;
        for (int k = 0; k < 4; ++k) { topi[b * 4 + k] = ti[k]; topw[b * 4 + k] = w[k] * inv; }
      }
      __syncthreads();
      // gather memory readout
      for (int i = tid; i < 1024; i += 256) {
        int b = i >> 7, d = i & 127;
        float acc = 0.0f;
        for (int k = 0; k < 4; ++k) {
          int idx = topi[b * 4 + k];
          float v = (idx < 16) ? (float)wv[((l * 8 + b) * 16 + idx) * 128 + d]
                               : sem_v[(l * 64 + (idx - 16)) * 128 + d];
          acc += topw[b * 4 + k] * v;
        }
        memb[i] = acc;
      }
      __syncthreads();
      for (int i = tid; i < 1024; i += 256)  // A cols 256..383 = mem
        Ab[(i >> 7) * 384 + 256 + (i & 127)] = (_Float16)memb[i];
      __syncthreads();
      gemm8(WgP + l * 49152, Ab, 12, gb, bg + l * 128, lane, wave);
      gemm8(WcP + l * 49152, Ab, 12, cb, bc + l * 128, lane, wave);
      __syncthreads();
      // h = (1-g)*prev + g*c ; repack A cols 128..255 = h
      for (int i = tid; i < 1024; i += 256) {
        int b = i >> 7, d = i & 127;
        float g  = 1.0f / (1.0f + __expf(-gb[i]));
        float cc = tanhf(cb[i]);
        float hv = (1.0f - g) * prev[i] + g * cc;
        Ab[b * 384 + 128 + d] = (_Float16)hv;
      }
      // overlap: warm Wo/Wk/Wv for the tail of this layer
      prefetch_w(WoP + l * 32768, 32768, tid);
      prefetch_w(WkP + l * 16384, 16384, tid);
      prefetch_w(WvP + l * 16384, 16384, tid);
      __syncthreads();
      gemm8(WoP + l * 32768, Ab, 8, pb, bo + l * 128, lane, wave);  // proj
      __syncthreads();
      {  // h_out = LN(proj + cur), wave per row
        float x[4];
#pragma unroll
        for (int i = 0; i < 4; ++i) {
          int d = lane + i * 32;
          x[i] = pb[wave * 128 + d] + cur[wave * 128 + d];
        }
        float s  = wave_sum(x[0] + x[1] + x[2] + x[3]);
        float s2 = wave_sum(x[0] * x[0] + x[1] * x[1] + x[2] * x[2] + x[3] * x[3]);
        float mean = s * 0.0078125f;
        float var  = s2 * 0.0078125f - mean * mean;
        float rs = rsqrtf(var + 1e-5f);
#pragma unroll
        for (int i = 0; i < 4; ++i) {
          int d = lane + i * 32;
          float ho = (x[i] - mean) * rs * lng[l * 128 + d] + lnb[l * 128 + d];
          cur[wave * 128 + d] = ho;
          hid[l * 1024 + wave * 128 + d] = ho;
        }
      }
      __syncthreads();
      {  // write gate = sigmoid(h_out . Wwg + bwg)
        float p = 0.0f;
#pragma unroll
        for (int i = 0; i < 4; ++i) {
          int d = lane + i * 32;
          p += cur[wave * 128 + d] * Wwg[l * 128 + d];
        }
        p = wave_sum(p);
        if (lane == 0) gate[wave] = 1.0f / (1.0f + __expf(-(p + bwg[l])));
      }
      for (int i = tid; i < 1024; i += 256)  // A cols 0..127 = h_out
        Ab[(i >> 7) * 384 + (i & 127)] = (_Float16)cur[i];
      // overlap: warm next layer's Wq slice
      if (l < 2) prefetch_w(WqP + (l + 1) * 32768, 32768, tid);
      __syncthreads();
      gemm8(WkP + l * 16384, Ab, 4, nkb, bk + l * 128, lane, wave);
      gemm8(WvP + l * 16384, Ab, 4, nvb, bv + l * 128, lane, wave);
      __syncthreads();
      {  // one-hot slot update
        int slot = t & 15;
        for (int i = tid; i < 1024; i += 256) {
          int b = i >> 7, d = i & 127;
          float g = gate[b];
          int base = ((l * 8 + b) * 16 + slot) * 128 + d;
          float nk = tanhf(nkb[i]);
          float nv = tanhf(nvb[i]);
          wk[base] = (_Float16)((float)wk[base] * (1.0f - g) + nk * g);
          wv[base] = (_Float16)((float)wv[base] * (1.0f - g) + nv * g);
        }
      }
      __syncthreads();
    }
    {  // out_t = LN(cur; fng,fnb) -> hseq f16
      float x[4];
#pragma unroll
      for (int i = 0; i < 4; ++i) x[i] = cur[wave * 128 + lane + i * 32];
      float s  = wave_sum(x[0] + x[1] + x[2] + x[3]);
      float s2 = wave_sum(x[0] * x[0] + x[1] * x[1] + x[2] * x[2] + x[3] * x[3]);
      float mean = s * 0.0078125f;
      float var  = s2 * 0.0078125f - mean * mean;
      float rs = rsqrtf(var + 1e-5f);
#pragma unroll
      for (int i = 0; i < 4; ++i) {
        int d = lane + i * 32;
        float ho = (x[i] - mean) * rs * fng[d] + fnb[d];
        hseq[((wave << 8) + t) * 128 + d] = (_Float16)ho;
      }
    }
    __syncthreads();
  }
}

// ---------------------------------------------------------------------------
// K4: out[2048,32000] = hseq @ Whead. Block = M64 x N128, 8 waves, 4 M-subtiles.
// Output is 262 MB streaming fp32 (> L2) -> non-temporal stores.
// ---------------------------------------------------------------------------
__global__ void __launch_bounds__(256)
head_k(const _Float16* __restrict__ hseq, const _Float16* __restrict__ WhP,
       float* __restrict__ out) {
  __shared__ _Float16 As[64 * 128];
  int tid = threadIdx.x, lane = tid & 31, wave = tid >> 5;
  int nb = blockIdx.x;  // 0..249 (128 cols each)
  int mb = blockIdx.y;  // 0..31  (64 rows each)
  for (int i = tid; i < 1024; i += 256)  // 8192 halves, 16B vectors
    ((v4u*)As)[i] = ((const v4u*)(hseq + (long)mb * 8192))[i];
  __syncthreads();
  v8f acc[4] = {};
  int ntg = nb * 8 + wave;  // global 16-col tile 0..1999
  const _Float16* wp = WhP + (size_t)ntg * 4 * 512 + lane * 16;
  for (int kc = 0; kc < 4; ++kc) {
    v16h b = *(const v16h*)(wp + kc * 512);
#pragma unroll
    for (int mi = 0; mi < 4; ++mi) {
      const _Float16* arow = As + (mi * 16 + (lane & 15)) * 128 + kc * 32;
      v16h a;
#pragma unroll
      for (int j = 0; j < 16; ++j) a[j] = arow[kmap(lane, j)];
      acc[mi] = WMMA_F16(a, b, acc[mi]);
    }
  }
  int col = nb * 128 + wave * 16 + (lane & 15);
  int mbase = (lane & 16) ? 8 : 0;
#pragma unroll
  for (int mi = 0; mi < 4; ++mi)
#pragma unroll
    for (int r = 0; r < 8; ++r) {
      int row = mb * 64 + mi * 16 + mbase + r;
      __builtin_nontemporal_store(acc[mi][r], out + (long)row * 32000 + col);
    }
}

// ---------------------------------------------------------------------------
extern "C" void kernel_launch(void* const* d_in, const int* in_sizes, int n_in,
                              void* d_out, int out_size, void* d_ws, size_t ws_size,
                              hipStream_t stream) {
  const int*   ids   = (const int*)d_in[0];
  const float* emb   = (const float*)d_in[1];
  const float* pos   = (const float*)d_in[2];
  const float* Win   = (const float*)d_in[3];
  const float* b_in  = (const float*)d_in[4];
  const float* Wq    = (const float*)d_in[5];
  const float* Wg    = (const float*)d_in[6];
  const float* bg    = (const float*)d_in[7];
  const float* Wc    = (const float*)d_in[8];
  const float* bc    = (const float*)d_in[9];
  const float* Wo    = (const float*)d_in[10];
  const float* bo    = (const float*)d_in[11];
  const float* lng   = (const float*)d_in[12];
  const float* lnb   = (const float*)d_in[13];
  const float* Wk    = (const float*)d_in[14];
  const float* bk    = (const float*)d_in[15];
  const float* Wv    = (const float*)d_in[16];
  const float* bv    = (const float*)d_in[17];
  const float* Wwg   = (const float*)d_in[18];
  const float* bwg   = (const float*)d_in[19];
  const float* sem_k = (const float*)d_in[20];
  const float* sem_v = (const float*)d_in[21];
  const float* fng   = (const float*)d_in[22];
  const float* fnb   = (const float*)d_in[23];
  const float* Whead = (const float*)d_in[24];
  float* out = (float*)d_out;

  _Float16* wsh  = (_Float16*)d_ws;
  _Float16* xh   = wsh;                 // 262144 halves
  _Float16* hseq = xh + 262144;         // 262144
  _Float16* WinP = hseq + 262144;       // 12288
  _Float16* WqP  = WinP + 12288;        // 3*32768
  _Float16* WgP  = WqP + 98304;         // 3*49152
  _Float16* WcP  = WgP + 147456;        // 3*49152
  _Float16* WoP  = WcP + 147456;        // 3*32768
  _Float16* WkP  = WoP + 98304;         // 3*16384
  _Float16* WvP  = WkP + 49152;         // 3*16384
  _Float16* WhP  = WvP + 49152;         // 4096000

  auto launch_pack = [&](const float* src, _Float16* dst, int Kd, int N) {
    long total = (long)Kd * N;
    int blocks = (int)((total + 255) / 256);
    pack_wmma_b_k<<<dim3(blocks), dim3(256), 0, stream>>>(src, dst, Kd, N);
  };

  launch_pack(Win, WinP, 96, 128);
  for (int l = 0; l < 3; ++l) {
    launch_pack(Wq + (long)l * 256 * 128, WqP + l * 32768, 256, 128);
    launch_pack(Wg + (long)l * 384 * 128, WgP + l * 49152, 384, 128);
    launch_pack(Wc + (long)l * 384 * 128, WcP + l * 49152, 384, 128);
    launch_pack(Wo + (long)l * 256 * 128, WoP + l * 32768, 256, 128);
    launch_pack(Wk + (long)l * 128 * 128, WkP + l * 16384, 128, 128);
    launch_pack(Wv + (long)l * 128 * 128, WvP + l * 16384, 128, 128);
  }
  launch_pack(Whead, WhP, 128, 32000);

  embed_in_k<<<dim3(128), dim3(256), 0, stream>>>(ids, emb, pos, WinP, b_in, xh);

  const int smem = 305952;  // < 320 KB CDNA5 WGP LDS
  hipFuncSetAttribute(reinterpret_cast<const void*>(scan_k),
                      hipFuncAttributeMaxDynamicSharedMemorySize, smem);
  scan_k<<<dim3(1), dim3(256), smem, stream>>>(
      xh, WqP, WgP, WcP, WoP, WkP, WvP, bg, bc, bo, lng, lnb, bk, bv, Wwg, bwg,
      sem_k, sem_v, fng, fnb, hseq);

  head_k<<<dim3(250, 32), dim3(256), 0, stream>>>(hseq, WhP, out);

  (void)in_sizes; (void)n_in; (void)out_size; (void)ws_size;
}
